// TFN_8839042695323
// MI455X (gfx1250) — compile-verified, matrix operand order checked by reading, not run on hardware
//
#include <hip/hip_runtime.h>
#include <hip/hip_bf16.h>
#include <math.h>

typedef __attribute__((ext_vector_type(16))) _Float16 v16h;
typedef __attribute__((ext_vector_type(8)))  float    v8f;
typedef __attribute__((ext_vector_type(4)))  float    v4f;
typedef __attribute__((ext_vector_type(4)))  int      v4i;

#define BATCH 32
#define NPTS0 1024
#define PATCH 32
#define KD48  48
#define GSCALE 6.23832462504f  /* ln(2) * 9 */
#define BNEPS 1e-3f

/* ------------------------------------------------------------------ */
/* 1) kd-tree level sort: one workgroup per (batch, block). Bitonic    */
/* ------------------------------------------------------------------ */
__global__ void tfn_kdtree_level(const float* __restrict__ src, float* __restrict__ dst,
                                 int level) {
    const int nb = 1 << level;
    const int bs = NPTS0 >> level;
    const int b  = blockIdx.x / nb;
    const int blk = blockIdx.x - b * nb;
    const size_t base = (size_t)b * NPTS0 + (size_t)blk * bs;

    __shared__ float sx[NPTS0], sy[NPTS0], sz[NPTS0], skey[NPTS0];
    __shared__ int   spos[NPTS0];
    __shared__ float rmn[3][256], rmx[3][256];
    __shared__ int   sax;

    const int tid = threadIdx.x, nth = blockDim.x;

    for (int i = tid; i < bs; i += nth) {
        sx[i] = src[(base + i) * 3 + 0];
        sy[i] = src[(base + i) * 3 + 1];
        sz[i] = src[(base + i) * 3 + 2];
    }
    __syncthreads();

    float mn0 = 1e30f, mn1 = 1e30f, mn2 = 1e30f;
    float mx0 = -1e30f, mx1 = -1e30f, mx2 = -1e30f;
    for (int i = tid; i < bs; i += nth) {
        mn0 = fminf(mn0, sx[i]); mx0 = fmaxf(mx0, sx[i]);
        mn1 = fminf(mn1, sy[i]); mx1 = fmaxf(mx1, sy[i]);
        mn2 = fminf(mn2, sz[i]); mx2 = fmaxf(mx2, sz[i]);
    }
    rmn[0][tid] = mn0; rmn[1][tid] = mn1; rmn[2][tid] = mn2;
    rmx[0][tid] = mx0; rmx[1][tid] = mx1; rmx[2][tid] = mx2;
    __syncthreads();
    for (int s = nth >> 1; s > 0; s >>= 1) {
        if (tid < s) {
            for (int a = 0; a < 3; ++a) {
                rmn[a][tid] = fminf(rmn[a][tid], rmn[a][tid + s]);
                rmx[a][tid] = fmaxf(rmx[a][tid], rmx[a][tid + s]);
            }
        }
        __syncthreads();
    }
    if (tid == 0) {
        float s0 = rmx[0][0] - rmn[0][0];
        float s1 = rmx[1][0] - rmn[1][0];
        float s2 = rmx[2][0] - rmn[2][0];
        int ax = 0; float bv = s0;
        if (s1 > bv) { ax = 1; bv = s1; }
        if (s2 > bv) { ax = 2; }
        sax = ax;
    }
    __syncthreads();
    const int ax = sax;
    for (int i = tid; i < bs; i += nth) {
        skey[i] = (ax == 0) ? sx[i] : (ax == 1) ? sy[i] : sz[i];
        spos[i] = i;
    }
    __syncthreads();

    for (int size = 2; size <= bs; size <<= 1) {
        for (int stride = size >> 1; stride > 0; stride >>= 1) {
            for (int t = tid; t < (bs >> 1); t += nth) {
                int i = (t / stride) * (stride << 1) + (t % stride);
                int j = i + stride;
                bool up = ((i & size) == 0);
                float ki = skey[i], kj = skey[j];
                if ((ki > kj) == up) {
                    skey[i] = kj; skey[j] = ki;
                    int p = spos[i]; spos[i] = spos[j]; spos[j] = p;
                }
            }
            __syncthreads();
        }
    }

    for (int i = tid; i < bs; i += nth) {
        int p = spos[i];
        dst[(base + i) * 3 + 0] = sx[p];
        dst[(base + i) * 3 + 1] = sy[p];
        dst[(base + i) * 3 + 2] = sz[p];
    }
}

/* ------------------------------------------------------------------ */
/* 2) kd-tree average pooling                                          */
/* ------------------------------------------------------------------ */
__global__ void tfn_kd_pool(const float* __restrict__ src, float* __restrict__ dst,
                            int n_out, int pool) {
    int g = blockIdx.x * blockDim.x + threadIdx.x;
    int total = BATCH * n_out * 3;
    if (g >= total) return;
    int comp = g % 3, rest = g / 3;
    int o = rest % n_out, b = rest / n_out;
    float s = 0.f;
    for (int j = 0; j < pool; ++j)
        s += src[(((size_t)b * n_out + o) * pool + j) * 3 + comp];
    dst[((size_t)b * n_out + o) * 3 + comp] = s / (float)pool;
}

/* ------------------------------------------------------------------ */
/* 3) 32-NN grouping                                                   */
/* ------------------------------------------------------------------ */
__global__ void tfn_group_knn(const float* __restrict__ src, const float* __restrict__ tgt,
                              int Ns, int Nt, int* __restrict__ idx) {
    int g = blockIdx.x * blockDim.x + threadIdx.x;
    if (g >= BATCH * Nt) return;
    int b = g / Nt, t = g - b * Nt;
    const float* tp = tgt + ((size_t)b * Nt + t) * 3;
    float tx = tp[0], ty = tp[1], tz = tp[2];
    float bd[PATCH]; int bi[PATCH];
    for (int k = 0; k < PATCH; ++k) { bd[k] = 1e30f; bi[k] = 0; }
    for (int j = 0; j < Ns; ++j) {
        const float* s = src + ((size_t)b * Ns + j) * 3;
        float dx = tx - s[0], dy = ty - s[1], dz = tz - s[2];
        float d = dx * dx + dy * dy + dz * dz;
        if (d < bd[PATCH - 1]) {
            int k = PATCH - 1;
            while (k > 0 && bd[k - 1] > d) { bd[k] = bd[k - 1]; bi[k] = bi[k - 1]; --k; }
            bd[k] = d; bi[k] = j;
        }
    }
    for (int k = 0; k < PATCH; ++k) idx[(size_t)g * PATCH + k] = bi[k];
}

/* ------------------------------------------------------------------ */
/* 4) SH x Gaussian kernels, written directly in B-fragment layout:   */
/*    kernp[((bt*3 + ntile)*32 + lane)*16 + elem] (f16)                */
/* ------------------------------------------------------------------ */
__global__ void tfn_sh_gauss(const float* __restrict__ src, const float* __restrict__ tgt,
                             const int* __restrict__ idx, _Float16* __restrict__ kernp,
                             int Ns, int Nt, float radius) {
    int g = blockIdx.x * blockDim.x + threadIdx.x;
    if (g >= BATCH * Nt * PATCH) return;
    int p  = g & (PATCH - 1);
    int bt = g >> 5;
    int b = bt / Nt, t = bt - b * Nt;
    int si = idx[g];
    const float* s  = src + ((size_t)b * Ns + si) * 3;
    const float* tp = tgt + ((size_t)b * Nt + t) * 3;
    float px = (s[0] - tp[0]) / radius;
    float py = (s[1] - tp[1]) / radius;
    float pz = (s[2] - tp[2]) / radius;
    float d  = sqrtf(px * px + py * py + pz * pz + 1e-12f);
    float dm = fmaxf(d, 1e-6f);
    float x = px / dm, y = py / dm, z = pz / dm;
    float x2 = x * x, y2 = y * y, z2 = z * z;
    float sh[16];
    sh[0]  = 0.28209479177f;
    sh[1]  = 0.4886025119f * y;
    sh[2]  = 0.4886025119f * z;
    sh[3]  = 0.4886025119f * x;
    sh[4]  = 1.0925484306f * x * y;
    sh[5]  = 1.0925484306f * y * z;
    sh[6]  = 0.3153915653f * (3.0f * z2 - 1.0f);
    sh[7]  = 1.0925484306f * x * z;
    sh[8]  = 0.5462742153f * (x2 - y2);
    sh[9]  = 0.5900435899f * y * (3.0f * x2 - y2);
    sh[10] = 2.8906114426f * x * y * z;
    sh[11] = 0.4570457995f * y * (5.0f * z2 - 1.0f);
    sh[12] = 0.3731763326f * z * (5.0f * z2 - 3.0f);
    sh[13] = 0.4570457995f * x * (5.0f * z2 - 1.0f);
    sh[14] = 1.4453057213f * z * (x2 - y2);
    sh[15] = 0.5900435899f * x * (x2 - 3.0f * y2);

    float mask = (d <= 1.0f) ? 1.0f : 0.0f;
    float g0 = expf(-GSCALE * d * d) * mask;
    float d1 = d - 0.5f, d2 = d - 1.0f;
    float g1 = expf(-GSCALE * d1 * d1) * mask;
    float g2 = expf(-GSCALE * d2 * d2) * mask;
    float gs = g0 + g1 + g2 + 1e-6f;
    g0 /= gs; g1 /= gs; g2 /= gs;
    float gg[3] = {g0, g1, g2};

    const int lane_half = ((p >> 3) & 1) * 16;
    const int elem = (p & 7) + ((p >= 16) ? 8 : 0);
    #pragma unroll
    for (int l = 0; l < 16; ++l) {
        #pragma unroll
        for (int ss = 0; ss < 3; ++ss) {
            int dch = l * 3 + ss;
            int ntile = dch >> 4, lane = (dch & 15) + lane_half;
            kernp[(((size_t)bt * 3 + ntile) * 32 + lane) * 16 + elem] =
                (_Float16)(sh[l] * gg[ss]);
        }
    }
}

/* ------------------------------------------------------------------ */
/* 5) Patch einsum: M=C, K=P=32, N=48. One wave per (bt, mtile, nt).  */
/*    Full m-tiles: strength-reduced unguarded stores. Partial tile:  */
/*    branchless offset select into a dump region of the same buffer. */
/* ------------------------------------------------------------------ */
__global__ void tfn_einsum_wmma(const float* __restrict__ yprev, int Cprev, int layer0,
                                const float* __restrict__ pts,
                                const int* __restrict__ idx,
                                const _Float16* __restrict__ kernp,
                                float* __restrict__ out, size_t dump_off,
                                int Nt, int Ns, int C, int Kpad) {
    const int lane = threadIdx.x;          /* one wave */
    const int bt = blockIdx.x;
    const int mt = blockIdx.y, ntile = blockIdx.z;
    const int b = bt / Nt;
    const int row = lane & 15, half = lane >> 4;
    const int c = mt * 16 + row;

    /* patch indices for this lane's two K runs: 4 x int4 */
    const v4i* ipr = (const v4i*)(idx + (size_t)bt * PATCH);
    v4i i0 = ipr[half * 2], i1 = ipr[half * 2 + 1];
    v4i i2 = ipr[4 + half * 2], i3 = ipr[4 + half * 2 + 1];
    int ipv[16];
    #pragma unroll
    for (int e = 0; e < 4; ++e) {
        ipv[e] = i0[e]; ipv[4 + e] = i1[e]; ipv[8 + e] = i2[e]; ipv[12 + e] = i3[e];
    }

    float av[16];
    if (c < C) {
        if (c < Cprev) {
            if (layer0) {
                #pragma unroll
                for (int e = 0; e < 16; ++e) av[e] = 1.0f;
            } else {
                #pragma unroll
                for (int e = 0; e < 16; ++e)
                    av[e] = yprev[((size_t)b * Ns + ipv[e]) * Cprev + c];
            }
        } else {
            int comp = c - Cprev;                       /* yzx */
            int cm = (comp == 0) ? 1 : (comp == 1) ? 2 : 0;
            #pragma unroll
            for (int e = 0; e < 16; ++e)
                av[e] = pts[((size_t)b * Ns + ipv[e]) * 3 + cm];
        }
    } else {
        #pragma unroll
        for (int e = 0; e < 16; ++e) av[e] = 0.f;
    }

    v16h a;
    #pragma unroll
    for (int e = 0; e < 16; ++e) a[e] = (_Float16)av[e];

    v16h bf = *(const v16h*)(kernp + (((size_t)bt * 3 + ntile) * 32 + lane) * 16);

    v8f acc = {};
    acc = __builtin_amdgcn_wmma_f32_16x16x32_f16(false, a, false, bf,
                                                 (short)0, acc, false, false);

    const int dch = ntile * 16 + (lane & 15);
    if (mt * 16 + 16 <= C) {                    /* wave-uniform: full tile */
        float* q = out + (size_t)bt * Kpad
                       + (size_t)(mt * 16 + half * 8) * KD48 + dch;
        #pragma unroll
        for (int r = 0; r < 8; ++r) { *q = acc[r]; q += KD48; }
    } else {                                     /* partial tile (one per grid.y) */
        #pragma unroll
        for (int r = 0; r < 8; ++r) {
            int cr = mt * 16 + half * 8 + r;
            size_t o = (cr < C)
                ? ((size_t)bt * Kpad + (size_t)cr * KD48 + dch)
                : (dump_off + lane);
            out[o] = acc[r];
        }
    }
}

/* ------------------------------------------------------------------ */
/* 6) Weight pre-pack into per-lane B-fragment layout (f16, padded).  */
/* ------------------------------------------------------------------ */
__global__ void tfn_pack_w(const float* __restrict__ W, _Float16* __restrict__ Wp,
                           int K, int N, int Kpad, int Npad) {
    int g = blockIdx.x * blockDim.x + threadIdx.x;
    int total = (Kpad >> 5) * (Npad >> 4) * 32;
    if (g >= total) return;
    int lane = g & 31, tile = g >> 5;
    int ntiles = Npad >> 4;
    int kt = tile / ntiles, nt = tile - kt * ntiles;
    int col = nt * 16 + (lane & 15), half = lane >> 4;
    _Float16 frag[16];
    #pragma unroll
    for (int e = 0; e < 16; ++e) {
        int kk = (e < 8) ? (half * 8 + e) : (16 + half * 8 + (e - 8));
        int k = kt * 32 + kk;
        frag[e] = (k < K && col < N) ? (_Float16)W[(size_t)k * N + col] : (_Float16)0.f;
    }
    v16h* dst = (v16h*)(Wp + (size_t)g * 16);
    *dst = *(const v16h*)frag;
}

/* zero-fill the [K, Kpad) tail of each activation row */
__global__ void tfn_pad_fill(float* __restrict__ A, int M, int K, int Kpad) {
    int pad = Kpad - K;
    int g = blockIdx.x * blockDim.x + threadIdx.x;
    if (g >= M * pad) return;
    int r = g / pad, cc = K + (g - r * pad);
    A[(size_t)r * Kpad + cc] = 0.f;
}

/* ------------------------------------------------------------------ */
/* 7) Dense GEMM via WMMA. A f32 (M x Kpad, fully valid, M%16==0),    */
/*    Bp pre-packed f16 fragments. 4 waves/block along N.             */
/* ------------------------------------------------------------------ */
__global__ void tfn_gemm_wmma(const float* __restrict__ A, const _Float16* __restrict__ Bp,
                              const float* __restrict__ bias, float* __restrict__ Z,
                              float* __restrict__ dump, int M, int Kpad, int N, int Npad) {
    const int lane = threadIdx.x;                       /* 32 */
    const int mt = blockIdx.x;
    const int nt = blockIdx.y * blockDim.y + threadIdx.y;
    if (nt * 16 >= Npad) return;                        /* wave-uniform */
    const int row = lane & 15, half = lane >> 4;
    const int ntiles = Npad >> 4;

    const v4f* Arow = (const v4f*)(A + (size_t)(mt * 16 + row) * Kpad);
    const v16h* Bfr = (const v16h*)Bp;

    v8f acc = {};
    const int nkt = Kpad >> 5;
    #pragma unroll 4
    for (int kt = 0; kt < nkt; ++kt) {
        int q0 = kt * 8 + half * 2;          /* run1: floats [kt*32 + half*8, +8) */
        v4f a0 = Arow[q0], a1 = Arow[q0 + 1];
        v4f a2 = Arow[q0 + 4], a3 = Arow[q0 + 5];  /* run2: +16 floats */
        v16h a;
        #pragma unroll
        for (int e = 0; e < 4; ++e) {
            a[e]      = (_Float16)a0[e];
            a[4 + e]  = (_Float16)a1[e];
            a[8 + e]  = (_Float16)a2[e];
            a[12 + e] = (_Float16)a3[e];
        }
        v16h b = Bfr[((size_t)kt * ntiles + nt) * 32 + lane];
        acc = __builtin_amdgcn_wmma_f32_16x16x32_f16(false, a, false, b,
                                                     (short)0, acc, false, false);
    }
    const int n = nt * 16 + (lane & 15);
    const bool ok = n < N;
    float bv = bias[ok ? n : 0];
    float* q = ok ? (Z + (size_t)(mt * 16 + half * 8) * N + n) : (dump + lane);
    const int stride = ok ? N : 0;
    #pragma unroll
    for (int r = 0; r < 8; ++r) { *q = acc[r] + bv; q += stride; }
}

/* ------------------------------------------------------------------ */
/* 8) BatchNorm statistics + apply                                     */
/* ------------------------------------------------------------------ */
__global__ void tfn_bn_stats(const float* __restrict__ Z, int Msamp, int N,
                             float* __restrict__ mean, float* __restrict__ var) {
    int c = blockIdx.x * blockDim.x + threadIdx.x;
    if (c >= N) return;
    float s = 0.f, s2 = 0.f;
    for (int i = 0; i < Msamp; ++i) {
        float v = Z[(size_t)i * N + c];
        s += v; s2 += v * v;
    }
    float m = s / (float)Msamp;
    mean[c] = m;
    var[c] = s2 / (float)Msamp - m * m;
}

__global__ void tfn_bn_relu(const float* __restrict__ Z, const float* __restrict__ mean,
                            const float* __restrict__ var, const float* __restrict__ gmm,
                            const float* __restrict__ bet, float* __restrict__ out,
                            int total, int N) {
    int g = blockIdx.x * blockDim.x + threadIdx.x;
    if (g >= total) return;
    int c = g % N;
    float v = (Z[g] - mean[c]) * rsqrtf(var[c] + BNEPS) * gmm[c] + bet[c];
    out[g] = fmaxf(v, 0.f);
}

/* ------------------------------------------------------------------ */
/* 9) Max-pool (B,16,1024) -> (B,1024)                                 */
/* ------------------------------------------------------------------ */
__global__ void tfn_maxpool16(const float* __restrict__ Y, float* __restrict__ out) {
    int g = blockIdx.x * blockDim.x + threadIdx.x;
    if (g >= BATCH * 1024) return;
    int b = g / 1024, c = g - b * 1024;
    float m = -1e30f;
    for (int t = 0; t < 16; ++t)
        m = fmaxf(m, Y[((size_t)b * 16 + t) * 1024 + c]);
    out[g] = m;
}

/* ------------------------------------------------------------------ */
/* 10) Row softmax on (32, 40) logits                                  */
/* ------------------------------------------------------------------ */
__global__ void tfn_softmax(const float* __restrict__ Z, float* __restrict__ out, int NC) {
    int b = threadIdx.x;
    if (b >= BATCH) return;
    const float* r = Z + (size_t)b * NC;
    float m = -1e30f;
    for (int j = 0; j < NC; ++j) m = fmaxf(m, r[j]);
    float s = 0.f;
    for (int j = 0; j < NC; ++j) s += expf(r[j] - m);
    for (int j = 0; j < NC; ++j) out[(size_t)b * NC + j] = expf(r[j] - m) / s;
}

/* ================================================================== */
extern "C" void kernel_launch(void* const* d_in, const int* in_sizes, int n_in,
                              void* d_out, int out_size, void* d_ws, size_t ws_size,
                              hipStream_t stream) {
    const float* x_in = (const float*)d_in[0];
    const float* Wj[3]  = {(const float*)d_in[1], (const float*)d_in[5], (const float*)d_in[9]};
    const float* Bj[3]  = {(const float*)d_in[2], (const float*)d_in[6], (const float*)d_in[10]};
    const float* Gj[3]  = {(const float*)d_in[3], (const float*)d_in[7], (const float*)d_in[11]};
    const float* BEj[3] = {(const float*)d_in[4], (const float*)d_in[8], (const float*)d_in[12]};
    const float* wf1 = (const float*)d_in[13]; const float* bf1 = (const float*)d_in[14];
    const float* gf1 = (const float*)d_in[15]; const float* bef1 = (const float*)d_in[16];
    const float* wf2 = (const float*)d_in[17]; const float* bf2 = (const float*)d_in[18];
    const float* gf2 = (const float*)d_in[19]; const float* bef2 = (const float*)d_in[20];
    const float* wsm = (const float*)d_in[21]; const float* bsm = (const float*)d_in[22];
    float* outp = (float*)d_out;

    char* w = (char*)d_ws;
    size_t off = 0;
    auto carve = [&](size_t bytes) -> void* {
        void* p = (void*)(w + off);
        off = (off + bytes + 255) & ~(size_t)255;
        return p;
    };
    float* p0 = (float*)carve((size_t)BATCH * 1024 * 3 * 4);
    float* p1 = (float*)carve((size_t)BATCH * 256 * 3 * 4);
    float* p2 = (float*)carve((size_t)BATCH * 64 * 3 * 4);
    float* p3 = (float*)carve((size_t)BATCH * 16 * 3 * 4);
    int*   idxb = (int*)carve((size_t)BATCH * 256 * PATCH * 4);
    _Float16* kernp = (_Float16*)carve((size_t)BATCH * 256 * 3 * 32 * 16 * 2);
    float* ein = (float*)carve((size_t)2048 * 3232 * 4);           /* max M*Kpad */
    _Float16* wpack = (_Float16*)carve((size_t)12448 * 1024 * 2);  /* max Kpad*Npad */
    float* zb  = (float*)carve((size_t)8192 * 64 * 4 > (size_t)512 * 1024 * 4
                               ? (size_t)8192 * 64 * 4 : (size_t)512 * 1024 * 4);
    float* yb  = (float*)carve((size_t)BATCH * 256 * 64 * 4);
    float* meanb = (float*)carve(1024 * 4);
    float* varb  = (float*)carve(1024 * 4);
    float* pooled = (float*)carve((size_t)BATCH * 1024 * 4);
    float* fc1 = (float*)carve((size_t)BATCH * 512 * 4);
    float* fc2 = (float*)carve((size_t)BATCH * 256 * 4);
    float* dump = (float*)carve(4096);
    const size_t dump_off = (size_t)(dump - ein);   /* float offset of dump from ein */
    (void)ws_size; (void)n_in; (void)in_sizes; (void)out_size;

    /* 1) kd-tree ordering */
    for (int level = 0; level < 10; ++level) {
        const float* src = (level == 0) ? x_in : p0;
        tfn_kdtree_level<<<BATCH << level, 256, 0, stream>>>(src, p0, level);
    }

    /* 2) kd pooling */
    tfn_kd_pool<<<(BATCH * 256 * 3 + 255) / 256, 256, 0, stream>>>(p0, p1, 256, 4);
    tfn_kd_pool<<<(BATCH * 64 * 3 + 255) / 256, 256, 0, stream>>>(p1, p2, 64, 4);
    tfn_kd_pool<<<(BATCH * 16 * 3 + 255) / 256, 256, 0, stream>>>(p2, p3, 16, 4);

    const float* pts[4] = {p0, p1, p2, p3};
    const int   NtA[3]   = {256, 64, 16};
    const int   NsA[3]   = {1024, 256, 64};
    const int   CA[3]    = {4, 67, 259};
    const int   CpA[3]   = {1, 64, 256};
    const int   KA[3]    = {192, 3216, 12432};
    const int   KpadA[3] = {192, 3232, 12448};
    const int   DA[3]    = {64, 256, 1024};
    const float radA[3]  = {0.2f, 0.4f, 0.8f};

    const float* yprev = nullptr;
    for (int i = 0; i < 3; ++i) {
        int Nt = NtA[i], Ns = NsA[i], C = CA[i], Cp = CpA[i];
        int K = KA[i], Kpad = KpadA[i], D = DA[i];
        int M = BATCH * Nt;

        tfn_group_knn<<<(M + 127) / 128, 128, 0, stream>>>(pts[i], pts[i + 1], Ns, Nt, idxb);
        tfn_sh_gauss<<<(M * PATCH + 255) / 256, 256, 0, stream>>>(
            pts[i], pts[i + 1], idxb, kernp, Ns, Nt, radA[i]);

        if (Kpad > K)
            tfn_pad_fill<<<(M * (Kpad - K) + 255) / 256, 256, 0, stream>>>(ein, M, K, Kpad);

        dim3 eg(M, (C + 15) / 16, 3);
        tfn_einsum_wmma<<<eg, 32, 0, stream>>>(yprev, Cp, (i == 0) ? 1 : 0,
                                               pts[i], idxb, kernp, ein, dump_off,
                                               Nt, Ns, C, Kpad);

        int pw = (Kpad / 32) * (D / 16) * 32;
        tfn_pack_w<<<(pw + 255) / 256, 256, 0, stream>>>(Wj[i], wpack, K, D, Kpad, D);

        dim3 gg(M / 16, (D + 63) / 64);
        tfn_gemm_wmma<<<gg, dim3(32, 4), 0, stream>>>(ein, wpack, Bj[i], zb, dump,
                                                      M, Kpad, D, D);

        tfn_bn_stats<<<(D + 127) / 128, 128, 0, stream>>>(zb, M, D, meanb, varb);
        tfn_bn_relu<<<(M * D + 255) / 256, 256, 0, stream>>>(
            zb, meanb, varb, Gj[i], BEj[i], yb, M * D, D);
        yprev = yb;
    }

    /* global max pool */
    tfn_maxpool16<<<(BATCH * 1024 + 255) / 256, 256, 0, stream>>>(yb, pooled);

    /* FC1 1024->512 */
    tfn_pack_w<<<((1024 / 32) * (512 / 16) * 32 + 255) / 256, 256, 0, stream>>>(
        wf1, wpack, 1024, 512, 1024, 512);
    tfn_gemm_wmma<<<dim3(2, 8), dim3(32, 4), 0, stream>>>(pooled, wpack, bf1, zb, dump,
                                                          BATCH, 1024, 512, 512);
    tfn_bn_stats<<<(512 + 127) / 128, 128, 0, stream>>>(zb, BATCH, 512, meanb, varb);
    tfn_bn_relu<<<(BATCH * 512 + 255) / 256, 256, 0, stream>>>(
        zb, meanb, varb, gf1, bef1, fc1, BATCH * 512, 512);

    /* FC2 512->256 */
    tfn_pack_w<<<((512 / 32) * (256 / 16) * 32 + 255) / 256, 256, 0, stream>>>(
        wf2, wpack, 512, 256, 512, 256);
    tfn_gemm_wmma<<<dim3(2, 4), dim3(32, 4), 0, stream>>>(fc1, wpack, bf2, zb, dump,
                                                          BATCH, 512, 256, 256);
    tfn_bn_stats<<<(256 + 127) / 128, 128, 0, stream>>>(zb, BATCH, 256, meanb, varb);
    tfn_bn_relu<<<(BATCH * 256 + 255) / 256, 256, 0, stream>>>(
        zb, meanb, varb, gf2, bef2, fc2, BATCH * 256, 256);

    /* classifier 256->40 (Npad=48) + softmax */
    tfn_pack_w<<<((256 / 32) * (48 / 16) * 32 + 255) / 256, 256, 0, stream>>>(
        wsm, wpack, 256, 40, 256, 48);
    tfn_gemm_wmma<<<dim3(2, 1), dim3(32, 4), 0, stream>>>(fc2, wpack, bsm, zb, dump,
                                                          BATCH, 256, 40, 48);
    tfn_softmax<<<1, 32, 0, stream>>>(zb, outp, 40);
}
